// NN2_SoftMin_10007273799762
// MI455X (gfx1250) — compile-verified, matrix operand order checked by reading, not run on hardware
//
#include <hip/hip_runtime.h>

// CDNA5 / gfx1250: wave32, WMMA 16x16x32 f16 -> f32
typedef _Float16 v16h __attribute__((ext_vector_type(16)));
typedef _Float16 v8h  __attribute__((ext_vector_type(8)));
typedef float    v8f  __attribute__((ext_vector_type(8)));

#define TAU_MIN 0.02f
#define TAU_MAX 2.0f
#define LN_EPS  1e-5f

__global__ __launch_bounds__(256) void softmin_mlp_wmma(
    const float* __restrict__ neigh3,
    const float* __restrict__ ln_w, const float* __restrict__ ln_b,
    const float* __restrict__ w1,  const float* __restrict__ b1,
    const float* __restrict__ w2,  const float* __restrict__ b2,
    const float* __restrict__ log_tau,
    float* __restrict__ out, int N)
{
  // Per-wave scratch, double-buffered f16 activation tile so the two 16x16
  // tiles form independent dependency chains (fills WMMA->VALU hazard slots).
  // Row stride 24 halves = 48B: row start and row+8 both 16B-aligned for
  // ds_load_b128. Plus a [32][4] f32 logits buffer. 8 waves per block.
  __shared__ __attribute__((aligned(16))) _Float16 sH[8][2][16][24];
  __shared__ __attribute__((aligned(16))) float    sL[8][32][4];

  const int lane = threadIdx.x & 31;
  const int wave = threadIdx.x >> 5;
  const int col  = lane & 15;      // tile column owned by this lane (C/D layout)
  const int hi   = lane >> 4;      // 0: lanes 0-15, 1: lanes 16-31
  const bool lo_half = (hi == 0);

  // ---------------- uniform parameters (computed once per wave) --------------
  float tau = __expf(log_tau[0]);
  tau = fminf(fmaxf(tau, TAU_MIN), TAU_MAX);
  const float inv_tau = 1.0f / tau;
  const float lw0 = ln_w[0], lw1 = ln_w[1], lw2 = ln_w[2];
  const float lb0 = ln_b[0], lb1 = ln_b[1], lb2 = ln_b[2];

  // B1 = W1^T zero-padded to 32x16 (K=0..2 valid). 16-bit B layout mirrors the
  // A layout: lanes 0-15 hold K=0..7 in the first 4 VGPR pairs, lanes 16-31
  // hold K=8..15 -> all zero here.
  v16h B1 = {};
  if (lo_half) {
    B1[0] = (_Float16)w1[col * 3 + 0];
    B1[1] = (_Float16)w1[col * 3 + 1];
    B1[2] = (_Float16)w1[col * 3 + 2];
  }
  // B2 = W2^T zero-padded to 32x16 (K=0..15 valid, only output cols 0..2 real).
  v16h B2 = {};
  if (col < 3) {
    const float* wrow = w2 + col * 16 + hi * 8;   // w2 is [3][16]
#pragma unroll
    for (int k = 0; k < 8; ++k) B2[k] = (_Float16)wrow[k];
  }
  // Fold biases into the WMMA C operands (loop-invariant broadcasts).
  v8f C1, C2;
  {
    const float bv1 = b1[col];
    const float bv2 = (col < 3) ? b2[col] : 0.0f;
#pragma unroll
    for (int r = 0; r < 8; ++r) { C1[r] = bv1; C2[r] = bv2; }
  }

  const int wavesPerBlock = blockDim.x >> 5;
  const int totalWaves = (int)gridDim.x * wavesPerBlock;
  const int gw = (int)blockIdx.x * wavesPerBlock + wave;
  const int stride = totalWaves * 32;

  for (int chunk = gw * 32; chunk < N; chunk += stride) {
    const int idx  = chunk + lane;
    const int idxc = (idx < N) ? idx : (N - 1);     // clamp: keep EXEC full
    const float* p = neigh3 + idxc * 3;
    const float x0 = p[0], x1 = p[1], x2 = p[2];

    // ---- soft-min: -tau * logsumexp(-x/tau) ----
    const float y0 = -x0 * inv_tau, y1 = -x1 * inv_tau, y2 = -x2 * inv_tau;
    const float m  = fmaxf(y0, fmaxf(y1, y2));
    const float s  = __expf(y0 - m) + __expf(y1 - m) + __expf(y2 - m);
    const float best = -tau * (m + __logf(s));

    // ---- feats + layernorm(3) ----
    const float f0 = -fabsf(x0 - best), f1 = -fabsf(x1 - best), f2 = -fabsf(x2 - best);
    const float mu = (f0 + f1 + f2) * (1.0f / 3.0f);
    const float d0 = f0 - mu, d1 = f1 - mu, d2 = f2 - mu;
    const float var = (d0 * d0 + d1 * d1 + d2 * d2) * (1.0f / 3.0f);
    const float rs  = __frsqrt_rn(var + LN_EPS);
    const float h0 = d0 * rs * lw0 + lb0;
    const float h1 = d1 * rs * lw1 + lb1;
    const float h2 = d2 * rs * lw2 + lb2;

    // Other half's activations (tile 1 A-matrix comes from lanes 16..31).
    const float g0 = __shfl_xor(h0, 16, 32);
    const float g1 = __shfl_xor(h1, 16, 32);
    const float g2 = __shfl_xor(h2, 16, 32);

    // ---- pack both A tiles branchlessly (v_cndmask, no EXEC churn) ----
    // A layout (f16 16x32): lanes 0-15 carry K=0..7 in elements 0..7; upper
    // lanes carry K=8..15 which are zero-padding here.
    v16h A0 = {}, A1 = {};
    A0[0] = (_Float16)(lo_half ? h0 : 0.0f);
    A0[1] = (_Float16)(lo_half ? h1 : 0.0f);
    A0[2] = (_Float16)(lo_half ? h2 : 0.0f);
    A1[0] = (_Float16)(lo_half ? g0 : 0.0f);
    A1[1] = (_Float16)(lo_half ? g1 : 0.0f);
    A1[2] = (_Float16)(lo_half ? g2 : 0.0f);

    // ---- GEMM1 for both tiles: H = relu(A * W1^T + b1) ----
    v8f D1a = __builtin_amdgcn_wmma_f32_16x16x32_f16(
        false, A0, false, B1, (short)0, C1, false, false);
    v8f D1b = __builtin_amdgcn_wmma_f32_16x16x32_f16(
        false, A1, false, B1, (short)0, C1, false, false);

    // ReLU, narrow to f16, transpose both tiles through LDS (intra-wave:
    // DS ops are in-order within a wave, no barrier needed).
#pragma unroll
    for (int r = 0; r < 8; ++r) {
      sH[wave][0][hi * 8 + r][col] = (_Float16)fmaxf(D1a[r], 0.0f);
      sH[wave][1][hi * 8 + r][col] = (_Float16)fmaxf(D1b[r], 0.0f);
    }
    // Re-read in GEMM2 A-layout: lane<16 -> row=lane, K=0..7;
    // lane>=16 -> row=lane-16, K=8..15. One aligned b128 per lane per tile.
    const v8h r0 = *(const v8h*)&sH[wave][0][col][hi * 8];
    const v8h r1 = *(const v8h*)&sH[wave][1][col][hi * 8];
    v16h A2a = {}, A2b = {};
#pragma unroll
    for (int k = 0; k < 8; ++k) { A2a[k] = r0[k]; A2b[k] = r1[k]; }

    // ---- GEMM2 for both tiles: logits = H * W2^T + b2 ----
    v8f D2a = __builtin_amdgcn_wmma_f32_16x16x32_f16(
        false, A2a, false, B2, (short)0, C2, false, false);
    v8f D2b = __builtin_amdgcn_wmma_f32_16x16x32_f16(
        false, A2b, false, B2, (short)0, C2, false, false);

    // Scatter valid logit columns so each lane can read its own element.
    if (col < 3) {
#pragma unroll
      for (int r = 0; r < 8; ++r) {
        sL[wave][hi * 8 + r][col]      = D2a[r];
        sL[wave][16 + hi * 8 + r][col] = D2b[r];
      }
    }

    // ---- epilogue: softmax over 3 logits + weighted sum (per lane) ----
    const float l0 = sL[wave][lane][0];
    const float l1 = sL[wave][lane][1];
    const float l2 = sL[wave][lane][2];
    const float mm = fmaxf(l0, fmaxf(l1, l2));
    const float e0 = __expf(l0 - mm), e1 = __expf(l1 - mm), e2 = __expf(l2 - mm);
    const float inv = 1.0f / (e0 + e1 + e2);
    const float res = (x0 * e0 + x1 * e1 + x2 * e2) * inv;
    if (idx < N) out[idx] = res;
  }
}

extern "C" void kernel_launch(void* const* d_in, const int* in_sizes, int n_in,
                              void* d_out, int out_size, void* d_ws, size_t ws_size,
                              hipStream_t stream) {
  const float* neigh3  = (const float*)d_in[0];
  const float* ln_w    = (const float*)d_in[1];
  const float* ln_b    = (const float*)d_in[2];
  const float* w1      = (const float*)d_in[3];
  const float* b1      = (const float*)d_in[4];
  const float* w2      = (const float*)d_in[5];
  const float* b2      = (const float*)d_in[6];
  const float* log_tau = (const float*)d_in[7];
  float* out = (float*)d_out;

  const int N = out_size;                // one scalar out per element
  int blocks = 4096;                     // persistent-style grid-stride
  const int need = (N + 255) / 256;
  if (blocks > need) blocks = (need > 0) ? need : 1;

  softmin_mlp_wmma<<<dim3(blocks), dim3(256), 0, stream>>>(
      neigh3, ln_w, ln_b, w1, b1, w2, b2, log_tau, out, N);
}
